// SSNHead_644245094461
// MI455X (gfx1250) — matrix operand order repaired and checked by reading, not compiled, not run on hardware
//
#include <hip/hip_runtime.h>

// ---------------------------------------------------------------------------
// SSN head for MI455X (gfx1250, wave32).
//
// Roofline: 301 MB x-read dominates vs ~5.4 GFLOP => memory bound (~15us floor
// at 23.3 TB/s). Strategy: read x exactly ONCE with non-temporal loads (x >
// 192MB L2, zero reuse -> don't evict weights), pool into a 16x5120 bf16 tile
// in LDS (160 KB -- exploits CDNA5's 320KB/WGP LDS), then 6 waves each run a
// v_wmma_f32_16x16x32_bf16 K-loop against the L2-resident swizzled combined
// weight matrix (96x5120: 3*w_act in course slice | w_comp | w_reg | pad).
// fp32 accumulation over K=5120 keeps bf16 input error negligible.
// ---------------------------------------------------------------------------

typedef __attribute__((ext_vector_type(16))) __bf16 v16bf;
typedef __attribute__((ext_vector_type(8)))  float  v8f;

#define IN_CH       1024
#define T_SEG       9
#define KTOT        (5 * IN_CH)            // 5120
#define KCHUNKS     (KTOT / 32)            // 160 K-chunks of 32
#define NPAD        96                     // 81 outputs padded to 6 col-tiles
#define JTILES      (NPAD / 16)            // 6
#define TILE_ELEMS  512                    // 16x32 bf16 operand tile
#define A_LDS_BYTES (KCHUNKS * TILE_ELEMS * 2)  // 163840 B = 160 KB

__device__ __forceinline__ unsigned short f2bf(float f) {
    unsigned int u = __float_as_uint(f);
    u += 0x7fffu + ((u >> 16) & 1u);       // round-to-nearest-even
    return (unsigned short)(u >> 16);
}

// ---------------------------------------------------------------------------
// Kernel 1: build (a) combined weight matrix, bf16, pre-swizzled into a contiguous-k WMMA
// B-operand lane layout (16-bit B 32x16: lanes 0-15 hold K=0..15 for col=lane,
// lanes 16-31 hold K=16..31) so the GEMM loop does contiguous 1KB tile loads;
// (b) a 96-entry epilogue table (out base, row stride, bias) so the GEMM
// epilogue is branch-free.
// ---------------------------------------------------------------------------
__global__ void ssn_prep(const float* __restrict__ w_act,
                         const float* __restrict__ w_comp,
                         const float* __restrict__ w_reg,
                         const float* __restrict__ b_act,
                         const float* __restrict__ b_comp,
                         const float* __restrict__ b_reg,
                         unsigned short* __restrict__ Bw,
                         int* __restrict__ tBase,
                         int* __restrict__ tStride,
                         float* __restrict__ tBias,
                         int N) {
    int idx = blockIdx.x * blockDim.x + threadIdx.x;   // over 96*5120
    if (idx >= NPAD * KTOT) return;
    int r = idx / KTOT;          // output row 0..95
    int k = idx - r * KTOT;      // 0..5119

    float v = 0.0f;
    if (r < 21) {
        // activity uses course_feat = 3 * p1 (p1 lives in slice [1024,2048))
        if ((k >> 10) == 1) v = 3.0f * w_act[r * IN_CH + (k & (IN_CH - 1))];
    } else if (r < 41) {
        v = w_comp[(size_t)(r - 21) * KTOT + k];
    } else if (r < 81) {
        v = w_reg[(size_t)(r - 41) * KTOT + k];
    } // rows 81..95: zero padding

    int jt   = r >> 4;           // col-tile
    int ncol = r & 15;           // column within tile
    int kc   = k >> 5;           // K-chunk
    int kk   = k & 31;           // K within chunk
    int lane = ncol + ((kk >> 4) << 4);
    int pos  = kk & 15;
    Bw[((size_t)jt * KCHUNKS + kc) * TILE_ELEMS + lane * 16 + pos] = f2bf(v);

    // Epilogue table: out[tBase[jg] + n*tStride[jg]] = acc + tBias[jg]
    if (idx < NPAD) {
        int jg = idx, base, stride;
        float bias;
        if (jg < 21)      { base = jg;                 stride = 21; bias = b_act[jg];       }
        else if (jg < 41) { base = N * 21 + (jg - 21); stride = 20; bias = b_comp[jg - 21]; }
        else if (jg < 81) { base = N * 41 + (jg - 41); stride = 40; bias = b_reg[jg - 41];  }
        else              { base = 0;                  stride = 0;  bias = 0.0f;            }
        tBase[jg]   = base;
        tStride[jg] = stride;
        tBias[jg]   = bias;
    }
}

// ---------------------------------------------------------------------------
// Kernel 2: fused pooling + WMMA GEMM. One workgroup (192 thr = 6 wave32)
// per 16-sample tile. Phase 1: pool 16x9x1024 fp32 (non-temporal reads) ->
// 16x5120 bf16 in LDS, swizzled to a contiguous-k 16-bit A layout (lanes 0-15:
// K 0..15, lanes 16-31: K 16..31). Phase 2: wave j
// accumulates its 16x16 output tile over 160 chunks with
// v_wmma_f32_16x16x32_bf16, then does a table-driven branch-free scatter.
// ---------------------------------------------------------------------------
__global__ void ssn_fused(const float* __restrict__ x,
                          const float* __restrict__ sf,
                          const unsigned short* __restrict__ Bw,
                          const int* __restrict__ tBase,
                          const int* __restrict__ tStride,
                          const float* __restrict__ tBias,
                          float* __restrict__ out, int N) {
    extern __shared__ unsigned short sA[];   // 160 KB dynamic LDS
    const int tid = threadIdx.x;
    const int n0  = blockIdx.x * 16;

    // ---- Phase 1: pooling into swizzled bf16 LDS tile ----
    for (int idx = tid; idx < 16 * IN_CH; idx += blockDim.x) {
        const int s = idx >> 10;             // sample within tile
        const int c = idx & (IN_CH - 1);     // channel (coalesced)
        float p0 = 0.f, p1 = 0.f, p2 = 0.f, p3 = 0.f, p4 = 0.f;
        const int n = n0 + s;
        if (n < N) {
            const float* xp = x + ((size_t)n * T_SEG) * IN_CH + c;
            // x is streamed once and exceeds L2: non-temporal so the
            // weight matrix stays L2-resident for all 512 workgroups.
            float x0 = __builtin_nontemporal_load(xp + 0 * IN_CH);
            float x1 = __builtin_nontemporal_load(xp + 1 * IN_CH);
            float x2 = __builtin_nontemporal_load(xp + 2 * IN_CH);
            float x3 = __builtin_nontemporal_load(xp + 3 * IN_CH);
            float x4 = __builtin_nontemporal_load(xp + 4 * IN_CH);
            float x5 = __builtin_nontemporal_load(xp + 5 * IN_CH);
            float x6 = __builtin_nontemporal_load(xp + 6 * IN_CH);
            float x7 = __builtin_nontemporal_load(xp + 7 * IN_CH);
            float x8 = __builtin_nontemporal_load(xp + 8 * IN_CH);
            float s0 = sf[2 * (size_t)n + 0], s1 = sf[2 * (size_t)n + 1];
            p0 = 0.5f * (x0 + x1) * s0;                       // start * sf0
            p1 = 0.2f * (x2 + x3 + x4 + x5 + x6) * (1.f/3.f); // course_all/3
            p2 = 0.5f * (x2 + x3) * (1.f/3.f);                // mean(2:4)/3
            p3 = (x4 + x5 + x6) * (1.f/9.f);                  // mean(4:7)/3
            p4 = 0.5f * (x7 + x8) * s1;                       // end * sf1
        }
        float pv[5] = {p0, p1, p2, p3, p4};
        #pragma unroll
        for (int p = 0; p < 5; ++p) {
            int k    = p * IN_CH + c;
            int kc   = k >> 5;
            int kk   = k & 31;
            int lane = s + ((kk >> 4) << 4);
            int pos  = kk & 15;
            sA[(size_t)kc * TILE_ELEMS + lane * 16 + pos] = f2bf(pv[p]);
        }
    }
    __syncthreads();

    // ---- Phase 2: WMMA GEMM, wave j -> output col-tile j ----
    const int wave = tid >> 5;               // 0..5
    const int lane = tid & 31;

    const unsigned short* aPtr = sA + lane * 16;
    const unsigned short* bPtr = Bw + (size_t)wave * KCHUNKS * TILE_ELEMS
                                    + lane * 16;
    v8f acc = {};
    for (int kc = 0; kc < KCHUNKS; ++kc) {
        v16bf a = *(const v16bf*)(aPtr + (size_t)kc * TILE_ELEMS);
        v16bf b = *(const v16bf*)(bPtr + (size_t)kc * TILE_ELEMS);
        if (kc + 1 < KCHUNKS)                // keep next weight tile warm
            __builtin_prefetch(bPtr + (size_t)(kc + 1) * TILE_ELEMS, 0, 1);
        acc = __builtin_amdgcn_wmma_f32_16x16x32_bf16(
            false, a, false, b, (short)0, acc, false, false);
    }

    // ---- Epilogue: table-driven branch-free scatter (C/D layout:
    //      lane = col + 16*(row>=8), VGPR v = row within half) ----
    const int col     = lane & 15;
    const int rowbase = (lane >> 4) * 8;
    const int jg      = wave * 16 + col;     // global output index 0..95
    if (jg >= 81) return;                    // padding columns
    const int    base   = tBase[jg];
    const int    stride = tStride[jg];
    const float  bias   = tBias[jg];
    #pragma unroll
    for (int v = 0; v < 8; ++v) {
        const int n = n0 + rowbase + v;
        if (n >= N) continue;
        out[(size_t)base + (size_t)n * stride] = acc[v] + bias;
    }
}

extern "C" void kernel_launch(void* const* d_in, const int* in_sizes, int n_in,
                              void* d_out, int out_size, void* d_ws, size_t ws_size,
                              hipStream_t stream) {
    const float* x      = (const float*)d_in[0];
    const float* sf     = (const float*)d_in[1];
    const float* w_act  = (const float*)d_in[2];
    const float* b_act  = (const float*)d_in[3];
    const float* w_comp = (const float*)d_in[4];
    const float* b_comp = (const float*)d_in[5];
    const float* w_reg  = (const float*)d_in[6];
    const float* b_reg  = (const float*)d_in[7];
    float* out = (float*)d_out;

    const int N = in_sizes[1] / 2;           // scale_factors is [N,2]

    // ws layout: [0, 96*5120) bf16 swizzled weights (~0.98 MB, L2-resident),
    // then the 96-entry epilogue table (base/stride/bias).
    unsigned short* Bw = (unsigned short*)d_ws;
    char* tbl = (char*)d_ws + (size_t)NPAD * KTOT * sizeof(unsigned short);
    int*   tBase   = (int*)tbl;
    int*   tStride = tBase + NPAD;
    float* tBias   = (float*)(tStride + NPAD);

    {
        int total  = NPAD * KTOT;
        int blocks = (total + 255) / 256;
        ssn_prep<<<blocks, 256, 0, stream>>>(w_act, w_comp, w_reg,
                                             b_act, b_comp, b_reg,
                                             Bw, tBase, tStride, tBias, N);
    }
    {
        int blocks = (N + 15) / 16;
        ssn_fused<<<blocks, JTILES * 32, A_LDS_BYTES, stream>>>(
            x, sf, Bw, tBase, tStride, tBias, out, N);
    }
}
